// TemporalGNNPredictor_53420803228010
// MI455X (gfx1250) — matrix operand order repaired and checked by reading, not compiled, not run on hardware
//
#include <hip/hip_runtime.h>
#include <hip/hip_bf16.h>
#include <cmath>

// ---- problem constants (from reference) ----
#define TT   12
#define NN   20000
#define EE   320000
#define F_IN 128
#define HHH  256
#define GGG  256
#define CCC  10

typedef __attribute__((ext_vector_type(16))) __bf16 v16bf;
typedef __attribute__((ext_vector_type(8)))  __bf16 v8bf;
typedef __attribute__((ext_vector_type(8)))  float  v8f;

// ---------------- utility kernels ----------------
__global__ void zero_kernel(float* __restrict__ p, int n) {
    int i = blockIdx.x * blockDim.x + threadIdx.x;
    if (i < n) p[i] = 0.0f;
}

__global__ void deg_kernel(const int* __restrict__ col,
                           const float* __restrict__ w,
                           float* __restrict__ deg) {
    int e = blockIdx.x * blockDim.x + threadIdx.x;
    if (e < EE) atomicAdd(&deg[col[e]], w[e]);
}

__global__ void dinv_kernel(const float* __restrict__ deg,
                            float* __restrict__ dinv) {
    int i = blockIdx.x * blockDim.x + threadIdx.x;
    if (i < NN) dinv[i] = rsqrtf(deg[i] + 1.0f);   // +1 = self-loop weight
}

// f32 -> bf16 row-major copy (activations; K stays contiguous)
__global__ void cvt_bf16_kernel(const float* __restrict__ src,
                                __bf16* __restrict__ dst, int n) {
    int i = blockIdx.x * blockDim.x + threadIdx.x;
    if (i < n) dst[i] = (__bf16)src[i];
}

// W[K x 256] f32 -> Wt[256 x K] bf16 (so B fragments are contiguous along K)
__global__ void tpose_bf16_kernel(const float* __restrict__ W,
                                  __bf16* __restrict__ Wt, int K) {
    int i = blockIdx.x * blockDim.x + threadIdx.x;
    if (i >= K * HHH) return;
    int k = i / HHH;
    int n = i - k * HHH;
    Wt[(size_t)n * K + k] = (__bf16)W[i];
}

// ---------------- WMMA bf16 GEMM:  C[M x 256] = A[M x K] * B[K x 256] ----------------
// A is bf16 row-major [M x K]; Bt is bf16 [256 x K] (= B transposed).
// Block = 128 threads (4 waves); each wave computes one 16x16 output tile.
// All fragment loads are 16-byte global_load_b128; K loop fully unrolled.
template<int K>
__global__ __launch_bounds__(128)
void wmma_gemm_bf16(const __bf16* __restrict__ A,
                    const __bf16* __restrict__ Bt,
                    float* __restrict__ C,          // row stride = 256
                    int M) {
    const int lane = threadIdx.x & 31;
    const int wave = threadIdx.x >> 5;
    const int half = lane >> 4;     // 0: lanes 0-15, 1: lanes 16-31
    const int l16  = lane & 15;

    const int M0 = (blockIdx.x * 4 + wave) * 16;
    const int n0 = blockIdx.y * 16;

    // A row m only affects D row m: clamp the address (no OOB), no value masking.
    const int mrow   = M0 + l16;
    const int mclamp = mrow < M ? mrow : (M - 1);
    const __bf16* __restrict__ arow = A  + (size_t)mclamp * K;
    const __bf16* __restrict__ brow = Bt + (size_t)(n0 + l16) * K;

    v8f acc = {};

    #pragma unroll
    for (int k0 = 0; k0 < K; k0 += 32) {
        // A fragment (16-bit A 16x32 layout):
        //   half h: VGPR0-3 <- K = k0+8h+(0..7); VGPR4-7 <- K = k0+16+8h+(0..7)
        v8bf a0 = *(const v8bf*)(arow + k0 + half * 8);
        v8bf a1 = *(const v8bf*)(arow + k0 + 16 + half * 8);
        // B fragment (16-bit B 32x16 layout):
        //   half h holds column n0+l16, K = k0+8h+(0..7) then k0+16+8h+(0..7) (matches A)
        v8bf b0 = *(const v8bf*)(brow + k0 + half * 8);
        v8bf b1 = *(const v8bf*)(brow + k0 + 16 + half * 8);

        v16bf a = __builtin_shufflevector(a0, a1, 0,1,2,3,4,5,6,7,8,9,10,11,12,13,14,15);
        v16bf b = __builtin_shufflevector(b0, b1, 0,1,2,3,4,5,6,7,8,9,10,11,12,13,14,15);

        acc = __builtin_amdgcn_wmma_f32_16x16x32_bf16(
                  false, a, false, b, (short)0, acc, false, false);
    }

    // C/D layout: VGPR v -> row M0 + 8*half + v, col n0 + l16
    float* __restrict__ crow = C + (size_t)(M0 + half * 8) * HHH + n0 + l16;
    if (M0 + 16 <= M) {                       // full tile: branch-free store
        #pragma unroll
        for (int v = 0; v < 8; ++v) crow[(size_t)v * HHH] = acc[v];
    } else {
        #pragma unroll
        for (int v = 0; v < 8; ++v) {
            if (M0 + half * 8 + v < M) crow[(size_t)v * HHH] = acc[v];
        }
    }
}

// ---------------- edge scatter: agg[col] += dinv[row]*w*dinv[col] * t[row] ----------------
// 64 lanes per edge, float4 gather (coalesced), f32 atomics absorbed by L2.
__global__ void scatter_kernel(const int* __restrict__ rowp,
                               const int* __restrict__ colp,
                               const float* __restrict__ w,
                               const float* __restrict__ dinv,
                               const float* __restrict__ t,
                               float* __restrict__ agg) {
    long gid = (long)blockIdx.x * blockDim.x + threadIdx.x;
    int e = (int)(gid >> 6);
    int j = (int)(gid & 63);
    if (e >= EE) return;
    int r = rowp[e], c = colp[e];
    float coeff = dinv[r] * w[e] * dinv[c];
    const float4 v = *(const float4*)(t + (size_t)r * HHH + j * 4);
    float* dst = agg + (size_t)c * HHH + j * 4;
    atomicAdd(dst + 0, coeff * v.x);
    atomicAdd(dst + 1, coeff * v.y);
    atomicAdd(dst + 2, coeff * v.z);
    atomicAdd(dst + 3, coeff * v.w);
}

// ---------------- finalize: h = relu(agg + dinv^2 * t + bias) ----------------
__global__ void finalize_kernel(const float* __restrict__ agg,
                                const float* __restrict__ t,
                                const float* __restrict__ dinv,
                                const float* __restrict__ bias,
                                float* __restrict__ out) {
    int i = blockIdx.x * blockDim.x + threadIdx.x;
    if (i >= NN * HHH) return;
    int node = i >> 8;
    int f = i & 255;
    float di = dinv[node];
    float v = agg[i] + di * di * t[i] + bias[f];
    out[i] = v > 0.0f ? v : 0.0f;
}

// ---------------- mean pool over nodes (two-stage, coalesced) ----------------
__global__ __launch_bounds__(256)
void pool_partial(const float* __restrict__ h, float* __restrict__ partial) {
    int f = threadIdx.x;               // feature
    int blk = blockIdx.x;              // 128 row chunks
    const int chunk = (NN + 127) / 128;
    int r0 = blk * chunk;
    int r1 = r0 + chunk; if (r1 > NN) r1 = NN;
    float s = 0.0f;
    for (int r = r0; r < r1; ++r) s += h[(size_t)r * HHH + f];
    partial[blk * HHH + f] = s;
}

__global__ __launch_bounds__(256)
void pool_final(const float* __restrict__ partial, float* __restrict__ emb_t) {
    int f = threadIdx.x;
    float s = 0.0f;
    #pragma unroll 4
    for (int b = 0; b < 128; ++b) s += partial[b * HHH + f];
    emb_t[f] = s * (1.0f / (float)NN);
}

// ---------------- GRU over T steps + linear classifier head ----------------
__global__ __launch_bounds__(GGG)
void gru_head(const float* __restrict__ emb,   // [T,256]
              const float* __restrict__ Wih,   // [768,256]
              const float* __restrict__ Whh,   // [768,256]
              const float* __restrict__ bih,   // [768]
              const float* __restrict__ bhh,   // [768]
              const float* __restrict__ Wc,    // [256,10]
              const float* __restrict__ bc,    // [10]
              float* __restrict__ out) {       // [10]
    __shared__ float hs[GGG];
    __shared__ float xs[GGG];
    const int tid = threadIdx.x;
    hs[tid] = 0.0f;
    __syncthreads();

    const float* wr = Wih + (size_t)tid * GGG;
    const float* wz = Wih + (size_t)(GGG + tid) * GGG;
    const float* wn = Wih + (size_t)(2 * GGG + tid) * GGG;
    const float* ur = Whh + (size_t)tid * GGG;
    const float* uz = Whh + (size_t)(GGG + tid) * GGG;
    const float* un = Whh + (size_t)(2 * GGG + tid) * GGG;

    for (int t = 0; t < TT; ++t) {
        xs[tid] = emb[t * GGG + tid];
        __syncthreads();
        float gxr = 0.f, gxz = 0.f, gxn = 0.f, ghr = 0.f, ghz = 0.f, ghn = 0.f;
        #pragma unroll 4
        for (int k = 0; k < GGG; ++k) {
            float xk = xs[k], hk = hs[k];
            gxr += wr[k] * xk; gxz += wz[k] * xk; gxn += wn[k] * xk;
            ghr += ur[k] * hk; ghz += uz[k] * hk; ghn += un[k] * hk;
        }
        float r = 1.0f / (1.0f + __expf(-(gxr + bih[tid] + ghr + bhh[tid])));
        float z = 1.0f / (1.0f + __expf(-(gxz + bih[GGG + tid] + ghz + bhh[GGG + tid])));
        float n = tanhf(gxn + bih[2 * GGG + tid] + r * (ghn + bhh[2 * GGG + tid]));
        float hnew = (1.0f - z) * n + z * hs[tid];
        __syncthreads();
        hs[tid] = hnew;
        __syncthreads();
    }

    if (tid < CCC) {
        float s = bc[tid];
        for (int k = 0; k < GGG; ++k) s += hs[k] * Wc[k * CCC + tid];
        out[tid] = s;
    }
}

// ---------------- launch ----------------
extern "C" void kernel_launch(void* const* d_in, const int* in_sizes, int n_in,
                              void* d_out, int out_size, void* d_ws, size_t ws_size,
                              hipStream_t stream) {
    const float* x   = (const float*)d_in[0];   // [T,N,128]
    const int*   ei  = (const int*)  d_in[1];   // [T,2,E]
    const float* ea  = (const float*)d_in[2];   // [T,E]
    const float* W1  = (const float*)d_in[3];   // [128,256]
    const float* b1  = (const float*)d_in[4];
    const float* W2  = (const float*)d_in[5];   // [256,256]
    const float* b2  = (const float*)d_in[6];
    const float* Wih = (const float*)d_in[7];
    const float* Whh = (const float*)d_in[8];
    const float* bih = (const float*)d_in[9];
    const float* bhh = (const float*)d_in[10];
    const float* Wc  = (const float*)d_in[11];
    const float* bc  = (const float*)d_in[12];
    float* out = (float*)d_out;

    // workspace layout (float-granular; bf16 buffers occupy half-floats)
    float* ws      = (float*)d_ws;
    float* deg     = ws;                                   // N
    float* dinv    = deg  + NN;                            // N
    float* tbuf    = dinv + NN;                            // N*256 (transform output, f32)
    float* hbuf    = tbuf + (size_t)NN * HHH;              // N*256 (activations, f32)
    float* agg     = hbuf + (size_t)NN * HHH;              // N*256 (aggregation, f32)
    float* partial = agg  + (size_t)NN * HHH;              // 128*256
    float* emb     = partial + 128 * HHH;                  // T*256
    __bf16* xbf    = (__bf16*)(emb + TT * HHH);            // N*128 bf16
    __bf16* hbf    = xbf + (size_t)NN * F_IN;              // N*256 bf16
    __bf16* w1t    = hbf + (size_t)NN * HHH;               // 256*128 bf16 (transposed)
    __bf16* w2t    = w1t + HHH * F_IN;                     // 256*256 bf16 (transposed)

    const int NH = NN * HHH;
    const dim3 gemmGrid((NN + 63) / 64, HHH / 16);
    const int  scatBlocks = (EE * 64 + 255) / 256;

    // one-shot weight prep: f32 -> bf16, transposed so K is contiguous
    tpose_bf16_kernel<<<(F_IN * HHH + 255) / 256, 256, 0, stream>>>(W1, w1t, F_IN);
    tpose_bf16_kernel<<<(HHH * HHH + 255) / 256, 256, 0, stream>>>(W2, w2t, HHH);

    for (int t = 0; t < TT; ++t) {
        const float* xt   = x  + (size_t)t * NN * F_IN;
        const int*   rowp = ei + (size_t)t * 2 * EE;
        const int*   colp = rowp + EE;
        const float* wt   = ea + (size_t)t * EE;

        // degree normalization
        zero_kernel<<<(NN + 255) / 256, 256, 0, stream>>>(deg, NN);
        deg_kernel<<<(EE + 255) / 256, 256, 0, stream>>>(colp, wt, deg);
        dinv_kernel<<<(NN + 255) / 256, 256, 0, stream>>>(deg, dinv);

        // ---- GCN layer 1 ----
        cvt_bf16_kernel<<<(NN * F_IN + 255) / 256, 256, 0, stream>>>(xt, xbf, NN * F_IN);
        wmma_gemm_bf16<F_IN><<<gemmGrid, 128, 0, stream>>>(xbf, w1t, tbuf, NN);
        zero_kernel<<<(NH + 255) / 256, 256, 0, stream>>>(agg, NH);
        scatter_kernel<<<scatBlocks, 256, 0, stream>>>(rowp, colp, wt, dinv, tbuf, agg);
        finalize_kernel<<<(NH + 255) / 256, 256, 0, stream>>>(agg, tbuf, dinv, b1, hbuf);

        // ---- GCN layer 2 ----
        cvt_bf16_kernel<<<(NH + 255) / 256, 256, 0, stream>>>(hbuf, hbf, NH);
        wmma_gemm_bf16<HHH><<<gemmGrid, 128, 0, stream>>>(hbf, w2t, tbuf, NN);
        zero_kernel<<<(NH + 255) / 256, 256, 0, stream>>>(agg, NH);
        scatter_kernel<<<scatBlocks, 256, 0, stream>>>(rowp, colp, wt, dinv, tbuf, agg);
        finalize_kernel<<<(NH + 255) / 256, 256, 0, stream>>>(agg, tbuf, dinv, b2, hbuf);

        // ---- mean pool -> emb[t] ----
        pool_partial<<<128, 256, 0, stream>>>(hbuf, partial);
        pool_final<<<1, 256, 0, stream>>>(partial, emb + t * HHH);
    }

    // ---- GRU + classifier ----
    gru_head<<<1, GGG, 0, stream>>>(emb, Wih, Whh, bih, bhh, Wc, bc, out);
}
